// Se3AttentionHead_53188874993901
// MI455X (gfx1250) — matrix operand order repaired and checked by reading, not compiled, not run on hardware
//
#include <hip/hip_runtime.h>

#define N_NODES 10000
#define N_EDGES 160000
#define NWK 128           // k tensor-product weights per edge
#define NWV 320           // v tensor-product weights per edge
#define NWTOT 448
#define N_LAYERS 3

typedef __attribute__((ext_vector_type(16))) _Float16 v16h;
typedef __attribute__((ext_vector_type(8)))  float    v8f;

// single v_rcp_f32 (~1 ulp) instead of the IEEE div_scale/div_fixup sequence
__device__ __forceinline__ float fast_rcp(float x) { return __builtin_amdgcn_rcpf(x); }
__device__ __forceinline__ float silu(float x) {
  return x * fast_rcp(1.0f + __expf(-x));
}

// ---------------------------------------------------------------------------
// prep_b: reorder R2k/R2v (f32, row-major 64 x NC, per layer) into f16 WMMA
// B-matrix fragments for v_wmma_f32_16x16x32_f16.
// B fragment layout (wave32, 32x16 tile): lane holds column n = 16t+(lane&15);
// half-slot i holds K = 32s + (lane>>4)*16 + i  (contiguous k order, same as the A fragments).
// ---------------------------------------------------------------------------
__global__ __launch_bounds__(64) void prep_b(const float* __restrict__ R2k,
                                             const float* __restrict__ R2v,
                                             _Float16* __restrict__ Bk,
                                             _Float16* __restrict__ Bv) {
  const int PER_LAYER = 56 * 32;            // (16 Bk + 40 Bv fragments) x 32 lanes
  int tid = blockIdx.x * blockDim.x + threadIdx.x;
  if (tid >= N_LAYERS * PER_LAYER) return;
  int layer = tid / PER_LAYER;
  int rem   = tid % PER_LAYER;
  int fi    = rem / 32;
  int lane  = rem % 32;
  int n16   = lane & 15;
  int khalf = lane >> 4;
  const float* src; _Float16* dstp; int NC, t, s;
  if (fi < 16) {                            // R2k: 8 N-tiles x 2 K-steps
    t = fi >> 1; s = fi & 1; NC = NWK;
    src  = R2k + layer * 64 * NWK;
    dstp = Bk + ((layer * 16 + fi) * 32 + lane) * 16;
  } else {                                  // R2v: 20 N-tiles x 2 K-steps
    int fj = fi - 16; t = fj >> 1; s = fj & 1; NC = NWV;
    src  = R2v + layer * 64 * NWV;
    dstp = Bv + ((layer * 40 + fj) * 32 + lane) * 16;
  }
  int n = t * 16 + n16;
  v16h frag;
#pragma unroll
  for (int i = 0; i < 16; ++i) {
    int k = s * 32 + khalf * 16 + i;
    frag[i] = (_Float16)src[k * NC + n];
  }
  *(v16h*)dstp = frag;
}

// ---------------------------------------------------------------------------
// node_init: zero layer output accumulator, q = xs @ Wq / sqrt(8),
// seed segment-max m = 1.0 (reference: m = max(segmax(logits), 1.0)).
// ---------------------------------------------------------------------------
__global__ __launch_bounds__(128) void node_init(const float* __restrict__ x_in,
                                                 const float* __restrict__ WqL,
                                                 float* __restrict__ q,
                                                 float* __restrict__ mbuf,
                                                 float* __restrict__ x_out) {
  int n = blockIdx.x * blockDim.x + threadIdx.x;
  if (n >= N_NODES) return;
  float4 z = make_float4(0.f, 0.f, 0.f, 0.f);
  float4* xo = (float4*)(x_out + n * 32);
#pragma unroll
  for (int i = 0; i < 8; ++i) xo[i] = z;
  float xs[8];
#pragma unroll
  for (int i = 0; i < 8; ++i) xs[i] = x_in[n * 32 + i];
  const float rs8 = 0.35355339059327373f;   // 1/sqrt(8)
#pragma unroll
  for (int o = 0; o < 8; ++o) {
    float a = 0.f;
#pragma unroll
    for (int i = 0; i < 8; ++i) a += xs[i] * WqL[i * 8 + o];
    q[n * 8 + o] = a * rs8;
  }
  mbuf[n] = 1.0f;
}

// ---------------------------------------------------------------------------
// edge_pass1: per wave, 16 edges.
//  1) build A fragments (silu radial activations) in registers in the same
//     contiguous k order as B: lane l -> row m = l&15; K = 32s + (l>>4)*16 + i
//  2) 56x v_wmma_f32_16x16x32_f16 -> per-edge weight tile (16x448 f32) in LDS
//     (tile loops unrolled x4 to cap in-flight B fragments and stay <256 VGPRs)
//  3) equivariant tensor products (2 lanes/edge), write vs/vv + logits,
//     atomicMax logits into segment max (positive-float int-max trick).
// ---------------------------------------------------------------------------
__global__ __launch_bounds__(64) void edge_pass1(
    const float* __restrict__ x_in,
    const int* __restrict__ src, const int* __restrict__ dst,
    const float* __restrict__ ef, const float* __restrict__ dist,
    const float* __restrict__ R1kL, const float* __restrict__ R1vL,
    const v16h* __restrict__ BkL, const v16h* __restrict__ BvL,
    const float* __restrict__ q, float* __restrict__ mbuf,
    float* __restrict__ logitbuf, float* __restrict__ vsvv) {
  __shared__ float wlds[2][16][NWTOT];      // 57344 B: 2 waves x 16 edges x 448
  const int wid   = threadIdx.x >> 5;
  const int lane  = threadIdx.x & 31;
  const int m16   = lane & 15;
  const int khalf = lane >> 4;
  const int eb    = (blockIdx.x * 2 + wid) * 16;

  // ---- A fragments: h = silu(d * W1), two K-steps each for k- and v-MLPs
  const float d = dist[eb + m16];
  v16h Ak0, Ak1, Av0, Av1;
#pragma unroll
  for (int i = 0; i < 16; ++i) {
    const int r = i >> 1, hh = i & 1, rr = r & 3;
    const int kb = khalf * 16 + i;
    Ak0[i] = (_Float16)silu(d * R1kL[kb]);
    Ak1[i] = (_Float16)silu(d * R1kL[kb + 32]);
    Av0[i] = (_Float16)silu(d * R1vL[kb]);
    Av1[i] = (_Float16)silu(d * R1vL[kb + 32]);
  }

  float (* __restrict__ W)[NWTOT] = wlds[wid];

  // ---- GEMM: wk = H_k(16x64) x R2k(64x128)
#pragma unroll 4
  for (int t = 0; t < 8; ++t) {
    v8f c = {};
    v16h b0 = BkL[(t * 2 + 0) * 32 + lane];
    v16h b1 = BkL[(t * 2 + 1) * 32 + lane];
    c = __builtin_amdgcn_wmma_f32_16x16x32_f16(false, Ak0, false, b0, (short)0, c, false, false);
    c = __builtin_amdgcn_wmma_f32_16x16x32_f16(false, Ak1, false, b1, (short)0, c, false, false);
#pragma unroll
    for (int r = 0; r < 8; ++r) W[khalf * 8 + r][t * 16 + m16] = c[r];
  }
  // ---- GEMM: wv = H_v(16x64) x R2v(64x320)
#pragma unroll 4
  for (int t = 0; t < 20; ++t) {
    v8f c = {};
    v16h b0 = BvL[(t * 2 + 0) * 32 + lane];
    v16h b1 = BvL[(t * 2 + 1) * 32 + lane];
    c = __builtin_amdgcn_wmma_f32_16x16x32_f16(false, Av0, false, b0, (short)0, c, false, false);
    c = __builtin_amdgcn_wmma_f32_16x16x32_f16(false, Av1, false, b1, (short)0, c, false, false);
#pragma unroll
    for (int r = 0; r < 8; ++r) W[khalf * 8 + r][NWK + t * 16 + m16] = c[r];
  }
  __syncthreads();

  // ---- tensor products: lane (m16) handles o=0..3, lane (m16+16) o=4..7
  const int e  = eb + m16;
  const int o0 = khalf * 4;
  const float* __restrict__ We = W[m16];
  const int s_n = src[e], d_n = dst[e];
  const float4 efv = *(const float4*)(ef + e * 4);
  const float gs = efv.x, g0 = efv.y, g1 = efv.z, g2 = efv.w;
  float xa[32];
  const float4* xp = (const float4*)(x_in + s_n * 32);
#pragma unroll
  for (int i = 0; i < 8; ++i) {
    float4 p = xp[i];
    xa[4*i] = p.x; xa[4*i+1] = p.y; xa[4*i+2] = p.z; xa[4*i+3] = p.w;
  }
  const float rs8 = 0.35355339059327373f;   // 1/sqrt(8)
  const float rs2 = 0.7071067811865475f;    // 1/sqrt(2)
  const float rs3 = 0.5773502691896258f;    // 1/sqrt(3)
  float dot[8], cr[8][3], xg[8];
#pragma unroll
  for (int i = 0; i < 8; ++i) {
    const float v0 = xa[8 + i*3], v1 = xa[9 + i*3], v2 = xa[10 + i*3];
    dot[i]   = (v0*g0 + v1*g1 + v2*g2) * rs3;
    cr[i][0] = (v1*g2 - v2*g1) * rs2;
    cr[i][1] = (v2*g0 - v0*g2) * rs2;
    cr[i][2] = (v0*g1 - v1*g0) * rs2;
    xg[i] = xa[i] * gs;
  }
  float lpart = 0.0f;
#pragma unroll
  for (int oo = 0; oo < 4; ++oo) {
    const int o = o0 + oo;
    float ak = 0.f, bk = 0.f, av = 0.f, bv = 0.f, t3 = 0.f;
    float v40 = 0.f, v41 = 0.f, v42 = 0.f, v50 = 0.f, v51 = 0.f, v52 = 0.f;
#pragma unroll
    for (int i = 0; i < 8; ++i) {
      const int io = i * 8 + o;
      ak += We[io]        * xg[i];
      bk += We[64  + io]  * dot[i];
      av += We[128 + io]  * xg[i];
      bv += We[192 + io]  * dot[i];
      t3 += We[256 + io]  * xa[i];
      const float w4 = We[320 + io];
      v40 += w4 * xa[8 + i*3]; v41 += w4 * xa[9 + i*3]; v42 += w4 * xa[10 + i*3];
      const float w5 = We[384 + io];
      v50 += w5 * cr[i][0]; v51 += w5 * cr[i][1]; v52 += w5 * cr[i][2];
    }
    const float kk = (ak + bk) * rs8 * rs2;
    const float vs = (av + bv) * rs8 * rs2;
    t3 *= rs8;
    lpart += q[d_n * 8 + o] * kk;
    vsvv[e * 32 + o]           = vs;
    vsvv[e * 32 + 8 + o*3 + 0] = (t3 * g0 + (v40 * gs + v50) * rs8) * rs3;
    vsvv[e * 32 + 8 + o*3 + 1] = (t3 * g1 + (v41 * gs + v51) * rs8) * rs3;
    vsvv[e * 32 + 8 + o*3 + 2] = (t3 * g2 + (v42 * gs + v52) * rs8) * rs3;
  }
  const float logit = (lpart + __shfl_xor(lpart, 16, 32)) * rs8;
  if (khalf == 0) {
    logitbuf[e] = logit;
    if (logit > 1.0f)                      // m >= 1.0 > 0: int-max on float bits is exact
      atomicMax((int*)(mbuf + d_n), __float_as_int(logit));
  }
}

// ---------------------------------------------------------------------------
__global__ __launch_bounds__(256) void node_pass2(const float* __restrict__ mbuf,
                                                  float* __restrict__ denom) {
  int n = blockIdx.x * blockDim.x + threadIdx.x;
  if (n < N_NODES) denom[n] = __expf(1.0f - mbuf[n]);   // softmax denominator seed
}

__global__ __launch_bounds__(256) void edge_pass3(float* __restrict__ logitbuf,
                                                  const int* __restrict__ dst,
                                                  const float* __restrict__ mbuf,
                                                  float* __restrict__ denom) {
  int e = blockIdx.x * blockDim.x + threadIdx.x;
  if (e < N_EDGES) {
    int d_n = dst[e];
    float ex = __expf(logitbuf[e] - mbuf[d_n]);
    logitbuf[e] = ex;
    atomicAdd(denom + d_n, ex);
  }
}

__global__ __launch_bounds__(256) void edge_pass4(const float* __restrict__ expe,
                                                  const float* __restrict__ denom,
                                                  const float* __restrict__ vsvv,
                                                  const int* __restrict__ dst,
                                                  float* __restrict__ x_out) {
  int idx = blockIdx.x * blockDim.x + threadIdx.x;
  if (idx < N_EDGES * 32) {
    int e = idx >> 5, c = idx & 31;
    int d_n = dst[e];
    float att = expe[e] * fast_rcp(denom[d_n]);
    atomicAdd(x_out + d_n * 32 + c, att * vsvv[e * 32 + c]);
  }
}

// ---------------------------------------------------------------------------
// Workspace byte offsets (total ~24.3 MB)
#define OFF_A   0u          // layer buffer A: 10000*32 f32
#define OFF_B   1280000u    // layer buffer B
#define OFF_Q   2560000u    // q: 10000*8 f32
#define OFF_M   2880000u    // segment max: 10000 f32
#define OFF_D   2920000u    // softmax denom: 10000 f32
#define OFF_L   2960000u    // per-edge logit/expe: 160000 f32
#define OFF_V   3600000u    // per-edge vs/vv: 160000*32 f32
#define OFF_BK  24080000u   // f16 B fragments for R2k: 3*16*32*16 halves
#define OFF_BV  24129152u   // f16 B fragments for R2v: 3*40*32*16 halves

extern "C" void kernel_launch(void* const* d_in, const int* in_sizes, int n_in,
                              void* d_out, int out_size, void* d_ws, size_t ws_size,
                              hipStream_t stream) {
  (void)in_sizes; (void)n_in; (void)out_size; (void)ws_size;
  const int*   ei   = (const int*)d_in[0];
  const int*   src  = ei;
  const int*   dst  = ei + N_EDGES;
  const float* nf   = (const float*)d_in[1];
  const float* ef   = (const float*)d_in[2];
  const float* dist = (const float*)d_in[3];
  const float* Wq   = (const float*)d_in[4];
  const float* R1k  = (const float*)d_in[5];
  const float* R2k  = (const float*)d_in[6];
  const float* R1v  = (const float*)d_in[7];
  const float* R2v  = (const float*)d_in[8];

  char* ws = (char*)d_ws;
  float* bufA     = (float*)(ws + OFF_A);
  float* bufB     = (float*)(ws + OFF_B);
  float* qbuf     = (float*)(ws + OFF_Q);
  float* mbuf     = (float*)(ws + OFF_M);
  float* denom    = (float*)(ws + OFF_D);
  float* logitbuf = (float*)(ws + OFF_L);
  float* vsvv     = (float*)(ws + OFF_V);
  _Float16* Bk    = (_Float16*)(ws + OFF_BK);
  _Float16* Bv    = (_Float16*)(ws + OFF_BV);

  prep_b<<<(N_LAYERS * 56 * 32 + 63) / 64, 64, 0, stream>>>(R2k, R2v, Bk, Bv);

  const float* xin = nf;
  for (int l = 0; l < N_LAYERS; ++l) {
    float* xout = (l == 0) ? bufA : (l == 1) ? bufB : (float*)d_out;
    node_init<<<(N_NODES + 127) / 128, 128, 0, stream>>>(xin, Wq + l * 64, qbuf, mbuf, xout);
    edge_pass1<<<N_EDGES / 32, 64, 0, stream>>>(
        xin, src, dst, ef, dist, R1k + l * 64, R1v + l * 64,
        (const v16h*)(Bk + (size_t)l * 16 * 32 * 16),
        (const v16h*)(Bv + (size_t)l * 40 * 32 * 16),
        qbuf, mbuf, logitbuf, vsvv);
    node_pass2<<<(N_NODES + 255) / 256, 256, 0, stream>>>(mbuf, denom);
    edge_pass3<<<(N_EDGES + 255) / 256, 256, 0, stream>>>(logitbuf, dst, mbuf, denom);
    edge_pass4<<<(N_EDGES * 32) / 256, 256, 0, stream>>>(logitbuf, denom, vsvv, dst, xout);
    xin = xout;
  }
}
